// Attention_52286931861800
// MI455X (gfx1250) — matrix operand order repaired and checked by reading, not compiled, not run on hardware
//
#include <hip/hip_runtime.h>
#include <hip/hip_bf16.h>

#define B_SZ 4
#define T_SZ 2048
#define C_SZ 2048
#define H_SZ 16
#define D_SZ 128

typedef __attribute__((ext_vector_type(16))) __bf16 v16bf;
typedef __attribute__((ext_vector_type(8)))  float  v8f;

union FragU {
    v16bf v;
    uint4 q[2];
};

// A-matrix 16x32 bf16 layout: lane row = lane%16, half = lane/16
// element e: k = k0 + 16*half + e  (16 contiguous)
__device__ __forceinline__ v16bf load_frag_a(const __bf16* __restrict__ row, int k0, int half) {
    FragU f;
    f.q[0] = *reinterpret_cast<const uint4*>(row + k0 + 16 * half);
    f.q[1] = *reinterpret_cast<const uint4*>(row + k0 + 16 * half + 8);
    return f.v;
}

// B-matrix 32x16 bf16 layout: lane col = lane%16, half = lane/16
// element e: k = k0 + 16*half + e  (16 contiguous)
__device__ __forceinline__ v16bf load_frag_b(const __bf16* __restrict__ row, int k0, int half) {
    FragU f;
    const uint4* p = reinterpret_cast<const uint4*>(row + k0 + 16 * half);
    f.q[0] = p[0];
    f.q[1] = p[1];
    return f.v;
}

// ---- CDNA5 async global->LDS copy (ASYNCcnt-tracked), per-lane 16B
__device__ __forceinline__ void async_load_b128(unsigned int lds_byte_off, const void* gaddr) {
    unsigned long long ga = (unsigned long long)(size_t)gaddr;
    asm volatile("global_load_async_to_lds_b128 %0, %1, off"
                 :: "v"(lds_byte_off), "v"(ga) : "memory");
}
__device__ __forceinline__ void wait_async0() {
#if defined(__has_builtin) && __has_builtin(__builtin_amdgcn_s_wait_asynccnt)
    __builtin_amdgcn_s_wait_asynccnt(0);
#else
    asm volatile("s_wait_asynccnt 0" ::: "memory");
#endif
}

// fast, overflow-safe tanh: tanh(x) = sign(x) * (1 - e^(-2|x|)) / (1 + e^(-2|x|))
__device__ __forceinline__ float fast_tanh(float x) {
    float t = __expf(-2.0f * fabsf(x));
    float r = (1.0f - t) / (1.0f + t);
    return copysignf(r, x);
}

// ---------------------------------------------------------------- casts
__global__ __launch_bounds__(256) void cast_f32_bf16(const float* __restrict__ in,
                                                     __bf16* __restrict__ out, int n4) {
    int i = blockIdx.x * 256 + threadIdx.x;
    if (i < n4) {
        float4 v = reinterpret_cast<const float4*>(in)[i];
        __bf16* o = out + 4 * (size_t)i;
        o[0] = (__bf16)v.x; o[1] = (__bf16)v.y; o[2] = (__bf16)v.z; o[3] = (__bf16)v.w;
    }
}

// in: R x C row-major f32  ->  outT: C x R row-major bf16
__global__ __launch_bounds__(256) void cast_transpose(const float* __restrict__ in,
                                                      __bf16* __restrict__ outT,
                                                      int R, int C) {
    __shared__ float tile[32][33];
    int tx = threadIdx.x & 31, ty = threadIdx.x >> 5;
    int c0 = blockIdx.x * 32, r0 = blockIdx.y * 32;
    for (int i = 0; i < 32; i += 8)
        tile[ty + i][tx] = in[(size_t)(r0 + ty + i) * C + c0 + tx];
    __syncthreads();
    for (int i = 0; i < 32; i += 8)
        outT[(size_t)(c0 + ty + i) * R + r0 + tx] = (__bf16)tile[tx][ty + i];
}

// ---------------------------------------------------------------- GEMM
// C[M,N] = A[M,K] * BT[N,K]^T ; 128x128 tile per block, 8 waves (2M x 4N),
// each wave: 64x32 = 4x2 WMMA accumulators. A/B re-reads hit the 192MB L2.
template <bool OUT_F32>
__global__ __launch_bounds__(256) void gemm_bf16(const __bf16* __restrict__ A,
                                                 const __bf16* __restrict__ BT,
                                                 void* __restrict__ Cout,
                                                 int M, int N, int K) {
    const int tiles_n = N >> 7;
    int bm = blockIdx.x / tiles_n, bn = blockIdx.x % tiles_n;
    int wave = threadIdx.x >> 5, lane = threadIdx.x & 31;
    int wm = wave >> 2, wn = wave & 3;
    int row0 = bm * 128 + wm * 64;
    int col0 = bn * 128 + wn * 32;
    int lm = lane & 15, half = lane >> 4;

    v8f acc[4][2] = {};
    const __bf16* arow[4];
    const __bf16* brow[2];
#pragma unroll
    for (int i = 0; i < 4; i++) arow[i] = A + (size_t)(row0 + i * 16 + lm) * K;
#pragma unroll
    for (int j = 0; j < 2; j++) brow[j] = BT + (size_t)(col0 + j * 16 + lm) * K;

    for (int k0 = 0; k0 < K; k0 += 32) {
        v16bf af[4], bf[2];
#pragma unroll
        for (int i = 0; i < 4; i++) af[i] = load_frag_a(arow[i], k0, half);
#pragma unroll
        for (int j = 0; j < 2; j++) bf[j] = load_frag_b(brow[j], k0, half);
#pragma unroll
        for (int i = 0; i < 4; i++)
#pragma unroll
            for (int j = 0; j < 2; j++)
                acc[i][j] = __builtin_amdgcn_wmma_f32_16x16x32_bf16(
                    false, af[i], false, bf[j], (short)0, acc[i][j], false, false);
    }
    // C layout: vgpr r -> row r + 8*half, col = lane%16
#pragma unroll
    for (int i = 0; i < 4; i++)
#pragma unroll
        for (int j = 0; j < 2; j++) {
            int r0 = row0 + i * 16 + 8 * half;
            int c = col0 + j * 16 + lm;
#pragma unroll
            for (int r = 0; r < 8; r++) {
                float v = acc[i][j][r];
                if (OUT_F32)
                    reinterpret_cast<float*>(Cout)[(size_t)(r0 + r) * N + c] = v;
                else
                    reinterpret_cast<__bf16*>(Cout)[(size_t)(r0 + r) * N + c] = (__bf16)v;
            }
        }
}

// ---------------------------------------------------------------- RoPE (in place on q,k heads)
// qkv: [B*T, 48, 128] bf16 ; heads 0..15 = q, 16..31 = k
__global__ __launch_bounds__(64) void rope_kernel(__bf16* __restrict__ qkv) {
    int bt = blockIdx.x;
    int t = bt & (T_SZ - 1);
    int head = blockIdx.y;  // 0..31
    int d = threadIdx.x;    // 0..63
    __bf16* p = qkv + ((size_t)bt * 48 + head) * D_SZ;
    float a = (float)p[d], b = (float)p[d + 64];
    float inv_ts = __powf(10000.0f, -(float)d * (1.0f / 64.0f));
    float s, c;
    sincosf((float)t * inv_ts, &s, &c);
    p[d]      = (__bf16)(a * c - b * s);
    p[d + 64] = (__bf16)(b * c + a * s);
}

// ---------------------------------------------------------------- flash attention
// grid: (T/128, B*H), block 256 (8 waves, each owns 16 q rows)
#define KT_STRIDE 136  // pad: rows land on distinct bank groups (272B = 68 dwords, 68%64=4)
__global__ __launch_bounds__(256) void flash_attn(const __bf16* __restrict__ qkv,
                                                  __bf16* __restrict__ out) {
    __shared__ __bf16 kt[32][KT_STRIDE];  // K tile, row-major, padded   (8704 B)
    __shared__ __bf16 vt[128][40];        // V tile transposed [d][t_local], pad 8
    __shared__ __bf16 pbuf[8][16][40];    // wave-private P staging

    int q0 = blockIdx.x * 128;
    int b = blockIdx.y >> 4, h = blockIdx.y & 15;
    int wave = threadIdx.x >> 5, lane = threadIdx.x & 31;
    int lm = lane & 15, half = lane >> 4;
    int wbase = q0 + wave * 16;

    const size_t rs = 48 * D_SZ;
    const __bf16* qbase = qkv + (size_t)b * T_SZ * rs + (size_t)h * D_SZ;
    const __bf16* kbase = qbase + (size_t)16 * D_SZ;
    const __bf16* vbase = qbase + (size_t)32 * D_SZ;

    // Q fragments (A role), kept for whole kernel
    v16bf qf[4];
    {
        const __bf16* qrow = qbase + (size_t)(wbase + lm) * rs;
#pragma unroll
        for (int f = 0; f < 4; f++) qf[f] = load_frag_a(qrow, f * 32, half);
    }

    v8f oacc[8] = {};
    float m_r[8], l_r[8];
#pragma unroll
    for (int r = 0; r < 8; r++) { m_r[r] = -3.0e38f; l_r[r] = 0.0f; }

    const float scale = 0.08838834764831845f;  // 1/sqrt(128)
    const int kv_end = q0 + 128;

    // per-thread staging indices: 256 threads cover a 32x128 tile in 16B chunks
    int srow = threadIdx.x >> 3;          // 0..31
    int scol = (threadIdx.x & 7) * 16;    // 0..112
    unsigned int kt_off = (unsigned int)(size_t)&kt[srow][scol];

    for (int kv = 0; kv < kv_end; kv += 32) {
        // async-stage K tile [kv..kv+32) x 128 verbatim into padded LDS
        async_load_b128(kt_off, kbase + (size_t)(kv + srow) * rs + scol);

        // cooperative transpose-stage of V tile [kv..kv+32) x 128 into vt[d][tl]
        {
            const __bf16* vrow = vbase + (size_t)(kv + srow) * rs + scol;
            uint4 w0 = *reinterpret_cast<const uint4*>(vrow);
            uint4 w1 = *reinterpret_cast<const uint4*>(vrow + 8);
            const __bf16* e0 = reinterpret_cast<const __bf16*>(&w0);
            const __bf16* e1 = reinterpret_cast<const __bf16*>(&w1);
#pragma unroll
            for (int i = 0; i < 8; i++) vt[scol + i][srow] = e0[i];
#pragma unroll
            for (int i = 0; i < 8; i++) vt[scol + 8 + i][srow] = e1[i];
        }
        wait_async0();
        __syncthreads();

        if (kv <= wbase) {  // tile contributes to this wave's rows
            // S = Q * K^T : two 16x16 tiles covering 32 kv columns; K from LDS
            v8f sacc[2] = {};
#pragma unroll
            for (int j = 0; j < 2; j++) {
                const __bf16* krow = &kt[j * 16 + lm][0];
#pragma unroll
                for (int f = 0; f < 4; f++) {
                    FragU kf;
                    const uint4* kp = reinterpret_cast<const uint4*>(krow + f * 32 + 16 * half);
                    kf.q[0] = kp[0];
                    kf.q[1] = kp[1];
                    sacc[j] = __builtin_amdgcn_wmma_f32_16x16x32_bf16(
                        false, qf[f], false, kf.v, (short)0, sacc[j], false, false);
                }
            }
            // softcap + causal mask + online softmax (per row r+8*half)
            float alpha[8];
#pragma unroll
            for (int r = 0; r < 8; r++) {
                int trow = wbase + r + 8 * half;
                float s0 = 50.0f * fast_tanh(sacc[0][r] * (scale * 0.02f));
                float s1 = 50.0f * fast_tanh(sacc[1][r] * (scale * 0.02f));
                if (kv + lm > trow)      s0 = -3.0e38f;
                if (kv + 16 + lm > trow) s1 = -3.0e38f;
                float mx = fmaxf(s0, s1);
#pragma unroll
                for (int off = 8; off >= 1; off >>= 1)
                    mx = fmaxf(mx, __shfl_xor(mx, off, 32));
                float mn = fmaxf(m_r[r], mx);
                float al = __expf(m_r[r] - mn);
                float p0 = __expf(s0 - mn);
                float p1 = __expf(s1 - mn);
                sacc[0][r] = p0; sacc[1][r] = p1;
                float sum = p0 + p1;
#pragma unroll
                for (int off = 8; off >= 1; off >>= 1)
                    sum += __shfl_xor(sum, off, 32);
                l_r[r] = l_r[r] * al + sum;
                m_r[r] = mn;
                alpha[r] = al;
            }
            // rescale running output
#pragma unroll
            for (int j = 0; j < 8; j++)
#pragma unroll
                for (int r = 0; r < 8; r++) oacc[j][r] *= alpha[r];

            // P: C layout -> A layout via wave-private LDS
#pragma unroll
            for (int j2 = 0; j2 < 2; j2++)
#pragma unroll
                for (int r = 0; r < 8; r++)
                    pbuf[wave][r + 8 * half][j2 * 16 + lm] = (__bf16)sacc[j2][r];

            v16bf pf;
            {
                const __bf16* prow = &pbuf[wave][lm][0];
                FragU f;
                f.q[0] = *reinterpret_cast<const uint4*>(prow + 16 * half);
                f.q[1] = *reinterpret_cast<const uint4*>(prow + 16 * half + 8);
                pf = f.v;
            }
            // O += P * V  (V from transposed LDS: B[k][n] = vt[16j+n][k])
#pragma unroll
            for (int j = 0; j < 8; j++) {
                const __bf16* vrow = &vt[j * 16 + lm][0];
                FragU f;
                const uint4* pp = reinterpret_cast<const uint4*>(vrow + 16 * half);
                f.q[0] = pp[0];
                f.q[1] = pp[1];
                oacc[j] = __builtin_amdgcn_wmma_f32_16x16x32_bf16(
                    false, pf, false, f.v, (short)0, oacc[j], false, false);
            }
        }
        __syncthreads();
    }

    // write out[b, t, h*128 + d] bf16
    float invl[8];
#pragma unroll
    for (int r = 0; r < 8; r++) invl[r] = __frcp_rn(l_r[r]);
#pragma unroll
    for (int j = 0; j < 8; j++)
#pragma unroll
        for (int r = 0; r < 8; r++) {
            int trow = wbase + r + 8 * half;
            float v = oacc[j][r] * invl[r];
            out[((size_t)b * T_SZ + trow) * C_SZ + h * D_SZ + j * 16 + lm] = (__bf16)v;
        }
}

// ---------------------------------------------------------------- launch
extern "C" void kernel_launch(void* const* d_in, const int* in_sizes, int n_in,
                              void* d_out, int out_size, void* d_ws, size_t ws_size,
                              hipStream_t stream) {
    const float* x     = (const float*)d_in[0];
    // d_in[1] = mask (causal, computed analytically)
    const float* w_qkv = (const float*)d_in[2];
    const float* w_out = (const float*)d_in[3];
    float* out_f = (float*)d_out;

    char* ws = (char*)d_ws;
    size_t o = 0;
    __bf16* xb    = (__bf16*)(ws + o); o += (size_t)B_SZ * T_SZ * C_SZ * 2;        // 33.5 MB
    __bf16* wqkvT = (__bf16*)(ws + o); o += (size_t)C_SZ * 3 * C_SZ * 2;           // 25.2 MB
    __bf16* woutT = (__bf16*)(ws + o); o += (size_t)C_SZ * C_SZ * 2;               // 8.4 MB
    __bf16* qkv   = (__bf16*)(ws + o); o += (size_t)B_SZ * T_SZ * 3 * C_SZ * 2;    // 100.7 MB
    __bf16* aout  = (__bf16*)(ws + o);                                             // 33.5 MB

    const int M = B_SZ * T_SZ;       // 8192
    const int N1 = 3 * C_SZ;         // 6144
    const int K = C_SZ;              // 2048

    // casts
    cast_f32_bf16<<<(M * K / 4 + 255) / 256, 256, 0, stream>>>(x, xb, M * K / 4);
    cast_transpose<<<dim3(N1 / 32, K / 32), 256, 0, stream>>>(w_qkv, wqkvT, K, N1);
    cast_transpose<<<dim3(C_SZ / 32, K / 32), 256, 0, stream>>>(w_out, woutT, K, C_SZ);

    // QKV projection
    gemm_bf16<false><<<(M / 128) * (N1 / 128), 256, 0, stream>>>(xb, wqkvT, qkv, M, N1, K);

    // RoPE in place on q,k heads
    rope_kernel<<<dim3(M, 32), 64, 0, stream>>>(qkv);

    // flash attention
    flash_attn<<<dim3(T_SZ / 128, B_SZ * H_SZ), 256, 0, stream>>>(qkv, aout);

    // output projection -> f32
    gemm_bf16<true><<<(M / 128) * (C_SZ / 128), 256, 0, stream>>>(aout, woutT, out_f, M, C_SZ, K);
}
